// MultiHeadAttention_558345748575
// MI455X (gfx1250) — matrix operand order repaired and checked by reading, not compiled, not run on hardware
//
#include <hip/hip_runtime.h>
#include <hip/hip_bf16.h>

// ---------------------------------------------------------------------------
// Types for CDNA5 WMMA (wave32): A/B = 16 bf16 per lane, C/D = 8 f32 per lane
// ---------------------------------------------------------------------------
typedef __bf16 bf16;
typedef __attribute__((ext_vector_type(16))) bf16  v16bf;
typedef __attribute__((ext_vector_type(8)))  float v8f;

union FragU { v16bf v; uint4 u[2]; };

// A fragment: 16x32 bf16, A stored row-major [M][lda] (K contiguous).
// ISA layout: lanes 0-15 M=0..15; lane half h: elems 0..7 -> K=kb+8h..+7,
// elems 8..15 -> K=kb+16+8h..+23  (two 16-byte loads).
__device__ __forceinline__ v16bf load_a_frag(const bf16* A, int lda, int row0,
                                             int kb, int lane) {
  const int m = row0 + (lane & 15);
  const int half = lane >> 4;
  const bf16* p = A + (long)m * lda + kb + 8 * half;
  FragU t;
  t.u[0] = *(const uint4*)(p);
  t.u[1] = *(const uint4*)(p + 16);
  return t.v;
}

// B fragment: 32x16 bf16, B supplied TRANSPOSED: Bt[n][k] (K contiguous).
// ISA layout: lane = n + 16*half, elems 0..7 -> K=kb+8h..+7, elems 8..15 -> K=kb+16+8h..+23.
__device__ __forceinline__ v16bf load_b_frag(const bf16* Bt, int ldb, int col0,
                                             int kb, int lane) {
  const int n = col0 + (lane & 15);
  const int half = lane >> 4;
  const bf16* p = Bt + (long)n * ldb + kb + 8 * half;
  FragU t;
  t.u[0] = *(const uint4*)(p);
  t.u[1] = *(const uint4*)(p + 16);
  return t.v;
}

__device__ __forceinline__ v8f wmma_bf16(v16bf a, v16bf b, v8f c) {
  return __builtin_amdgcn_wmma_f32_16x16x32_bf16(
      /*neg_a=*/false, a, /*neg_b=*/false, b,
      /*c_mod=*/(short)0, c, /*reuse_a=*/false, /*reuse_b=*/false);
}

// ---------------------------------------------------------------------------
// Elementwise prep kernels
// ---------------------------------------------------------------------------

// W [K][N] fp32 row-major  ->  Wt [N][K] bf16 (K contiguous for B fragments)
__global__ void tp_bf16_kernel(const float* __restrict__ W,
                               bf16* __restrict__ Wt, int K, int N) {
  long idx = (long)blockIdx.x * blockDim.x + threadIdx.x;
  long total = (long)K * N;
  if (idx >= total) return;
  int k = (int)(idx / N);
  int n = (int)(idx - (long)k * N);
  Wt[(long)n * K + k] = (bf16)W[idx];
}

// x [8192][512] fp32 -> bf16 into h buffer columns [0:512), ldo = 1024
__global__ void xconv_kernel(const float* __restrict__ x,
                             bf16* __restrict__ hbuf) {
  long idx = (long)blockIdx.x * blockDim.x + threadIdx.x;
  if (idx >= (long)8192 * 512) return;
  int row = (int)(idx >> 9);
  int col = (int)(idx & 511);
  hbuf[(long)row * 1024 + col] = (bf16)x[idx];
}

// ---------------------------------------------------------------------------
// Generic bf16 WMMA GEMM: C = act(A @ B + bias)
//   Wave tile: 32(M) x 64(N)  (2 A-frags x 4 B-frags = 8 WMMA per k-step)
//   Workgroup: 8 waves -> 256 rows.  grid.x = M/256, grid.y = N/64.
//   K-loop double-buffered in steps of 64 (K must be a multiple of 64).
// out modes: 0 = bf16 [m][ldo], 1 = f32 [m][ldo],
//            2 = cts/ngh scatter: m=b*1024+i, n=h*64+l -> [b][h][i][l] bf16
//            3 = vT scatter (N=64): -> [b][s][1024] bf16
// ---------------------------------------------------------------------------
__device__ __forceinline__ void load_ab(const bf16* A, int lda, int tileM,
                                        const bf16* Bt, int ldb, int tileN,
                                        int kb, int lane, v16bf a[2],
                                        v16bf b[4]) {
  a[0] = load_a_frag(A, lda, tileM, kb, lane);
  a[1] = load_a_frag(A, lda, tileM + 16, kb, lane);
#pragma unroll
  for (int sN = 0; sN < 4; ++sN)
    b[sN] = load_b_frag(Bt, ldb, tileN + sN * 16, kb, lane);
}

__device__ __forceinline__ void mma_set(const v16bf a[2], const v16bf b[4],
                                        v8f acc[2][4]) {
#pragma unroll
  for (int mi = 0; mi < 2; ++mi)
#pragma unroll
    for (int sN = 0; sN < 4; ++sN)
      acc[mi][sN] = wmma_bf16(a[mi], b[sN], acc[mi][sN]);
}

__global__ __launch_bounds__(256) void gemm_bf16_kernel(
    const bf16* __restrict__ A, int lda, const bf16* __restrict__ Bt, int ldb,
    const float* __restrict__ bias, int K, bf16* __restrict__ outb,
    float* __restrict__ outf, int ldo, int mode, int relu) {
  const int wave = threadIdx.x >> 5;
  const int lane = threadIdx.x & 31;
  const int half = lane >> 4;
  const int lan = lane & 15;
  const int tileM = blockIdx.x * 256 + wave * 32;
  const int tileN = blockIdx.y * 64;

  v8f acc[2][4] = {};
  v16bf a0[2], b0[4], a1[2], b1[4];
  load_ab(A, lda, tileM, Bt, ldb, tileN, 0, lane, a0, b0);
  // ping-pong double buffer: step 64 = two WMMA k-steps of 32
  for (int kb = 0; kb + 64 <= K; kb += 64) {
    load_ab(A, lda, tileM, Bt, ldb, tileN, kb + 32, lane, a1, b1);
    mma_set(a0, b0, acc);
    if (kb + 64 < K)
      load_ab(A, lda, tileM, Bt, ldb, tileN, kb + 64, lane, a0, b0);
    mma_set(a1, b1, acc);
  }

#pragma unroll
  for (int mi = 0; mi < 2; ++mi)
#pragma unroll
    for (int sN = 0; sN < 4; ++sN) {
      const int n = tileN + sN * 16 + lan;
      const float bv = bias ? bias[n] : 0.0f;
#pragma unroll
      for (int r = 0; r < 8; ++r) {
        const int m = tileM + mi * 16 + r + 8 * half;
        float v = acc[mi][sN][r] + bv;
        if (relu) v = fmaxf(v, 0.0f);
        if (mode == 0) {
          outb[(long)m * ldo + n] = (bf16)v;
        } else if (mode == 1) {
          outf[(long)m * ldo + n] = v;
        } else if (mode == 2) {
          const int b = m >> 10, i = m & 1023, h = n >> 6, l = n & 63;
          outb[(((long)(b * 8 + h)) * 1024 + i) * 64 + l] = (bf16)v;
        } else {  // mode 3: vT[b][s][i]
          outb[((long)(m >> 10) * 64 + n) * 1024 + (m & 1023)] = (bf16)v;
        }
      }
    }
}

// ---------------------------------------------------------------------------
// Flash-attention per (b,h): Q=ngh[b,h] [1024][64], K=cts[b,h] [1024][64],
// V^T = vT[b] [64][1024]. Row-softmax over i with LOCAL (per-head) max;
// emits unnormalized y_un[b][h][j][64] plus mstat/zstat[b*8+h][j] for the
// cross-head joint-normalization combine.
// grid = (64, 8): blockIdx.x = b*8+h, blockIdx.y = j-block of 128.
// Each of 8 waves owns 16 query rows j; the Q fragments are loop-invariant
// and hoisted out of both i-loops.
// ---------------------------------------------------------------------------
__global__ __launch_bounds__(256) void attn_kernel(
    const bf16* __restrict__ cts, const bf16* __restrict__ ngh,
    const bf16* __restrict__ vT, float* __restrict__ y_un,
    float* __restrict__ mstat, float* __restrict__ zstat) {
  __shared__ bf16 lds_p[8 * 16 * 32];  // per-wave 16x32 exp(P) staging (8 KB)

  const int bh = blockIdx.x;
  const int wave = threadIdx.x >> 5;
  const int lane = threadIdx.x & 31;
  const int half = lane >> 4;
  const int lan = lane & 15;
  const int j0 = blockIdx.y * 128 + wave * 16;

  const bf16* Q = ngh + (long)bh * 1024 * 64;         // rows j, K=l
  const bf16* Kt = cts + (long)bh * 1024 * 64;        // rows i, K=l
  const bf16* Vb = vT + (long)(bh >> 3) * 64 * 1024;  // rows s, K=i
  bf16* myp = lds_p + wave * 512;

  // Q fragments for this wave's 16 j-rows: invariant across all i-tiles
  v16bf bq[2];
  bq[0] = load_b_frag(Q, 64, j0, 0, lane);
  bq[1] = load_b_frag(Q, 64, j0, 32, lane);

  // ---- pass 1: per-column-j (lane) running max over all i ----
  float rmax = -3.0e38f;
  for (int it = 0; it < 64; ++it) {
    v16bf ka0 = load_a_frag(Kt, 64, it * 16, 0, lane);
    v16bf ka1 = load_a_frag(Kt, 64, it * 16, 32, lane);
    v8f s = {};
    s = wmma_bf16(ka0, bq[0], s);
    s = wmma_bf16(ka1, bq[1], s);
#pragma unroll
    for (int r = 0; r < 8; ++r) rmax = fmaxf(rmax, s[r]);
  }
  const float m_j = fmaxf(rmax, __shfl_xor(rmax, 16, 32));

  // ---- pass 2: p = exp(s - m_j); accumulate Z and y = P^T @ V ----
  v8f yacc[4] = {};
  float zsum = 0.0f;
  for (int ic = 0; ic < 1024; ic += 32) {
#pragma unroll
    for (int sub = 0; sub < 2; ++sub) {
      v16bf ka0 = load_a_frag(Kt, 64, ic + sub * 16, 0, lane);
      v16bf ka1 = load_a_frag(Kt, 64, ic + sub * 16, 32, lane);
      v8f s = {};
      s = wmma_bf16(ka0, bq[0], s);
      s = wmma_bf16(ka1, bq[1], s);
      union { bf16 h[8]; uint4 u; } pk;
#pragma unroll
      for (int r = 0; r < 8; ++r) {
        float p = __expf(s[r] - m_j);
        zsum += p;
        pk.h[r] = (bf16)p;
      }
      // transpose through LDS: row j_local = lan, cols i_local consecutive
      *(uint4*)(myp + lan * 32 + sub * 16 + 8 * half) = pk.u;
    }
    asm volatile("s_wait_dscnt 0" ::: "memory");
    v16bf a = load_a_frag(myp, 32, 0, 0, lane);  // P'[j][i] 16x32
#pragma unroll
    for (int sN = 0; sN < 4; ++sN) {
      v16bf b = load_b_frag(Vb, 1024, sN * 16, ic, lane);  // V^T rows s
      yacc[sN] = wmma_bf16(a, b, yacc[sN]);
    }
  }

  const float z_j = zsum + __shfl_xor(zsum, 16, 32);
  if (half == 0) {
    mstat[(long)bh * 1024 + j0 + lan] = m_j;
    zstat[(long)bh * 1024 + j0 + lan] = z_j;
  }
#pragma unroll
  for (int sN = 0; sN < 4; ++sN)
#pragma unroll
    for (int r = 0; r < 8; ++r) {
      const int j = j0 + r + 8 * half;
      const int sc = sN * 16 + lan;
      y_un[((long)bh * 1024 + j) * 64 + sc] = yacc[sN][r];
    }
}

// ---------------------------------------------------------------------------
// Cross-head softmax combine: per (b,j): M = max_h m_h, Z = sum_h Z_h e^{m_h-M};
// ybf[b*1024+j][h*64+s] = y_un[b][h][j][s] * e^{m_h-M} / Z   (bf16, GEMM-ready)
// grid = 8192 blocks (b*1024+j), 64 threads = s
// ---------------------------------------------------------------------------
__global__ void combine_kernel(const float* __restrict__ y_un,
                               const float* __restrict__ mstat,
                               const float* __restrict__ zstat,
                               bf16* __restrict__ ybf) {
  const int bj = blockIdx.x;
  const int b = bj >> 10, j = bj & 1023;
  const int t = threadIdx.x;
  float mh[8];
  float M = -3.0e38f;
#pragma unroll
  for (int h = 0; h < 8; ++h) {
    mh[h] = mstat[(long)(b * 8 + h) * 1024 + j];
    M = fmaxf(M, mh[h]);
  }
  float Z = 0.0f;
#pragma unroll
  for (int h = 0; h < 8; ++h)
    Z += zstat[(long)(b * 8 + h) * 1024 + j] * __expf(mh[h] - M);
#pragma unroll
  for (int h = 0; h < 8; ++h) {
    const float f = __expf(mh[h] - M) / Z;
    ybf[(long)bj * 512 + h * 64 + t] =
        (bf16)(y_un[((long)(b * 8 + h) * 1024 + j) * 64 + t] * f);
  }
}

// ---------------------------------------------------------------------------
// Host-side orchestration
// ---------------------------------------------------------------------------
extern "C" void kernel_launch(void* const* d_in, const int* in_sizes, int n_in,
                              void* d_out, int out_size, void* d_ws,
                              size_t ws_size, hipStream_t stream) {
  const float* x     = (const float*)d_in[0];
  const float* W_cts = (const float*)d_in[1];
  const float* W_ngh = (const float*)d_in[2];
  const float* W_com = (const float*)d_in[3];
  const float* W_grp = (const float*)d_in[4];
  const float* W0 = (const float*)d_in[5];  const float* b0 = (const float*)d_in[6];
  const float* W1 = (const float*)d_in[7];  const float* b1 = (const float*)d_in[8];
  const float* W2 = (const float*)d_in[9];  const float* b2 = (const float*)d_in[10];
  const float* W3 = (const float*)d_in[11]; const float* b3 = (const float*)d_in[12];
  const float* W4 = (const float*)d_in[13]; const float* b4 = (const float*)d_in[14];
  float* out = (float*)d_out;

  char* cur = (char*)d_ws;
  auto carve = [&](size_t bytes) {
    char* p = cur;
    cur += (bytes + 255) & ~(size_t)255;
    return p;
  };
  bf16* hbuf  = (bf16*)carve((size_t)8192 * 1024 * 2);  // [x | y@Wgrp] bf16
  bf16* ctsB  = (bf16*)carve((size_t)8 * 8 * 1024 * 64 * 2);
  bf16* nghB  = (bf16*)carve((size_t)8 * 8 * 1024 * 64 * 2);
  bf16* vTb   = (bf16*)carve((size_t)8 * 64 * 1024 * 2);
  bf16* WctsT = (bf16*)carve((size_t)512 * 512 * 2);
  bf16* WnghT = (bf16*)carve((size_t)512 * 512 * 2);
  bf16* WcomT = (bf16*)carve((size_t)64 * 512 * 2);
  bf16* WgrpT = (bf16*)carve((size_t)512 * 512 * 2);
  bf16* W0T   = (bf16*)carve((size_t)256 * 1024 * 2);
  bf16* W1T   = (bf16*)carve((size_t)256 * 256 * 2);
  bf16* W2T   = (bf16*)carve((size_t)256 * 256 * 2);
  bf16* W3T   = (bf16*)carve((size_t)256 * 256 * 2);
  bf16* W4T   = (bf16*)carve((size_t)512 * 256 * 2);
  float* y_un = (float*)carve((size_t)8 * 8 * 1024 * 64 * 4);
  float* mst  = (float*)carve((size_t)64 * 1024 * 4);
  float* zst  = (float*)carve((size_t)64 * 1024 * 4);
  bf16* ybf   = (bf16*)carve((size_t)8192 * 512 * 2);
  bf16* t0    = (bf16*)carve((size_t)8192 * 256 * 2);
  bf16* t1    = (bf16*)carve((size_t)8192 * 256 * 2);

  // x -> bf16 into h[:, 0:512)
  xconv_kernel<<<(8192 * 512 + 255) / 256, 256, 0, stream>>>(x, hbuf);

  // weight transposes to bf16 K-contiguous
  auto tp = [&](const float* W, bf16* Wt, int K, int N) {
    long tot = (long)K * N;
    tp_bf16_kernel<<<(unsigned)((tot + 255) / 256), 256, 0, stream>>>(W, Wt, K, N);
  };
  tp(W_cts, WctsT, 512, 512);
  tp(W_ngh, WnghT, 512, 512);
  tp(W_com, WcomT, 512, 64);
  tp(W_grp, WgrpT, 512, 512);
  tp(W0, W0T, 1024, 256);
  tp(W1, W1T, 256, 256);
  tp(W2, W2T, 256, 256);
  tp(W3, W3T, 256, 256);
  tp(W4, W4T, 256, 512);

  // encoder GEMMs (A = x bf16 inside hbuf, lda=1024, K=512)
  gemm_bf16_kernel<<<dim3(32, 8), 256, 0, stream>>>(
      hbuf, 1024, WctsT, 512, nullptr, 512, ctsB, nullptr, 0, 2, 0);
  gemm_bf16_kernel<<<dim3(32, 8), 256, 0, stream>>>(
      hbuf, 1024, WnghT, 512, nullptr, 512, nghB, nullptr, 0, 2, 0);
  gemm_bf16_kernel<<<dim3(32, 1), 256, 0, stream>>>(
      hbuf, 1024, WcomT, 512, nullptr, 512, vTb, nullptr, 0, 3, 0);

  // flash attention (per-head local softmax stats) + cross-head combine
  attn_kernel<<<dim3(64, 8), 256, 0, stream>>>(ctsB, nghB, vTb, y_un, mst, zst);
  combine_kernel<<<8192, 64, 0, stream>>>(y_un, mst, zst, ybf);

  // y @ W_grp -> h[:, 512:1024)
  gemm_bf16_kernel<<<dim3(32, 8), 256, 0, stream>>>(
      ybf, 512, WgrpT, 512, nullptr, 512, hbuf + 512, nullptr, 1024, 0, 0);

  // MLP: 1024->256 relu, 256->256 relu x3, 256->512 (+bias) -> fp32 out
  gemm_bf16_kernel<<<dim3(32, 4), 256, 0, stream>>>(
      hbuf, 1024, W0T, 1024, b0, 1024, t0, nullptr, 256, 0, 1);
  gemm_bf16_kernel<<<dim3(32, 4), 256, 0, stream>>>(
      t0, 256, W1T, 256, b1, 256, t1, nullptr, 256, 0, 1);
  gemm_bf16_kernel<<<dim3(32, 4), 256, 0, stream>>>(
      t1, 256, W2T, 256, b2, 256, t0, nullptr, 256, 0, 1);
  gemm_bf16_kernel<<<dim3(32, 4), 256, 0, stream>>>(
      t0, 256, W3T, 256, b3, 256, t1, nullptr, 256, 0, 1);
  gemm_bf16_kernel<<<dim3(32, 8), 256, 0, stream>>>(
      t1, 256, W4T, 256, b4, 256, nullptr, out, 512, 1, 0);

  (void)in_sizes; (void)n_in; (void)out_size; (void)ws_size;
}
